// MultiHeadAttention_56916906606987
// MI455X (gfx1250) — matrix operand order repaired and checked by reading, not compiled, not run on hardware
//
#include <hip/hip_runtime.h>

// ---------------------------------------------------------------------------
// MI455X (gfx1250) multi-head attention with physical-coordinate RoPE.
// All matmuls run on v_wmma_f32_16x16x32_bf16 (fp32 accumulate).
// Round 3: fix async-builtin signature — global_load_async_to_lds_b128 takes
// (int4 AS1*, int4 AS3*, imm offset, imm cpol) per hipcc's diagnostic.
// ---------------------------------------------------------------------------

typedef __attribute__((ext_vector_type(16))) __bf16 v16bf;
typedef __attribute__((ext_vector_type(8)))  __bf16 v8bf;
typedef __attribute__((ext_vector_type(8)))  float  v8f;
typedef __attribute__((ext_vector_type(4)))  int    v4i;

#define BATCH 4
#define SEQ   2048
#define NH    16
#define DH    96
#define DM    1536   // NH*DH == D_MODEL

#define GLOBAL_AS __attribute__((address_space(1)))
#define SHARED_AS __attribute__((address_space(3)))

#if defined(__has_builtin)
#if __has_builtin(__builtin_amdgcn_global_load_async_to_lds_b128) && \
    __has_builtin(__builtin_amdgcn_s_wait_asynccnt)
#define USE_ASYNC_LDS 1
#endif
#endif
#ifndef USE_ASYNC_LDS
#define USE_ASYNC_LDS 0
#endif

static __device__ __forceinline__ v8f wmma_bf16(v16bf a, v16bf b, v8f c) {
  // D = A(16x32) * B(32x16) + C, fp32 accumulate
  return __builtin_amdgcn_wmma_f32_16x16x32_bf16(false, a, false, b, (short)0, c,
                                                 false, false);
}
static __device__ __forceinline__ v16bf cat8(v8bf lo, v8bf hi) {
  return __builtin_shufflevector(lo, hi, 0, 1, 2, 3, 4, 5, 6, 7,
                                 8, 9, 10, 11, 12, 13, 14, 15);
}

#if USE_ASYNC_LDS
static __device__ __forceinline__ void async_ld_b128(const void* g, void* l) {
  __builtin_amdgcn_global_load_async_to_lds_b128(
      (GLOBAL_AS v4i*)(v4i*)(void*)g, (SHARED_AS v4i*)(v4i*)l, 0, 0);
}
#endif

// ---------------------------------------------------------------------------
// Generic GEMM: C[M x N] = A[M x K] * W[N x K]^T  (torch Linear convention).
// fp32 in, bf16 WMMA compute, fp32 out. Block tile 128x128x32, 8 waves.
// Register double-buffer: next tile's global loads are in flight while the
// current tile's WMMAs execute; the s_wait_loadcnt lands only at the convert.
// ---------------------------------------------------------------------------
#define BM 128
#define BN 128
#define BK 32
#define BKP 40   // padded LDS row: 80B -> 20-bank stride, conflict-free b128 frags

__global__ __launch_bounds__(256) void gemm_xwt_kernel(
    const float* __restrict__ A, const float* __restrict__ W,
    float* __restrict__ C, int M, int N, int K) {
  __shared__ __bf16 Als[BM][BKP];
  __shared__ __bf16 Bls[BN][BKP];

  const int tid  = threadIdx.x;
  const int lane = tid & 31;
  const int w    = tid >> 5;   // 0..7
  const int wm   = w >> 1;     // 0..3 -> 32-row strip
  const int wn   = w & 1;      // 0..1 -> 64-col strip
  const int half = lane >> 4;  // 0/1
  const int lm   = lane & 15;
  const int m0   = blockIdx.y * BM;
  const int n0   = blockIdx.x * BN;

  // Per-thread staging slots: idx = tid + i*256 -> row idx/8, col (idx%8)*4
  int ldr[4], ldc[4];
#pragma unroll
  for (int i = 0; i < 4; ++i) {
    int idx = tid + i * 256;
    ldr[i] = idx >> 3;
    ldc[i] = (idx & 7) << 2;
  }

  v8f zero = {};
  v8f c[2][4];
#pragma unroll
  for (int mt = 0; mt < 2; ++mt)
#pragma unroll
    for (int nt = 0; nt < 4; ++nt) c[mt][nt] = zero;

  // Prologue: issue loads for k0 = 0.
  float4 va[4], vb[4];
#pragma unroll
  for (int i = 0; i < 4; ++i) {
    va[i] = *reinterpret_cast<const float4*>(&A[(size_t)(m0 + ldr[i]) * K + ldc[i]]);
    vb[i] = *reinterpret_cast<const float4*>(&W[(size_t)(n0 + ldr[i]) * K + ldc[i]]);
  }

  for (int k0 = 0; k0 < K; k0 += BK) {
    // --- convert staged registers -> bf16 LDS tiles ---
#pragma unroll
    for (int i = 0; i < 4; ++i) {
      __bf16* da = &Als[ldr[i]][ldc[i]];
      da[0] = (__bf16)va[i].x; da[1] = (__bf16)va[i].y;
      da[2] = (__bf16)va[i].z; da[3] = (__bf16)va[i].w;
      __bf16* db = &Bls[ldr[i]][ldc[i]];
      db[0] = (__bf16)vb[i].x; db[1] = (__bf16)vb[i].y;
      db[2] = (__bf16)vb[i].z; db[3] = (__bf16)vb[i].w;
    }
    __syncthreads();

    // --- issue next tile's global loads (overlap with WMMA compute) ---
    const bool more = (k0 + BK) < K;
    float4 na[4], nb[4];
    if (more) {
#pragma unroll
      for (int i = 0; i < 4; ++i) {
        na[i] = *reinterpret_cast<const float4*>(
            &A[(size_t)(m0 + ldr[i]) * K + k0 + BK + ldc[i]]);
        nb[i] = *reinterpret_cast<const float4*>(
            &W[(size_t)(n0 + ldr[i]) * K + k0 + BK + ldc[i]]);
      }
    }

    // --- fragment loads (ISA 7.12.2 layouts) + 8 WMMAs ---
    v16bf a[2], b[4];
#pragma unroll
    for (int mt = 0; mt < 2; ++mt) {
      int row  = wm * 32 + mt * 16 + lm;          // A: lane = row M
      v8bf lo = *reinterpret_cast<const v8bf*>(&Als[row][half * 8]);
      v8bf hi = *reinterpret_cast<const v8bf*>(&Als[row][16 + half * 8]);
      a[mt] = cat8(lo, hi);
    }
#pragma unroll
    for (int nt = 0; nt < 4; ++nt) {
      int row  = wn * 64 + nt * 16 + lm;          // B: lane = col N, K contiguous
      v8bf lo = *reinterpret_cast<const v8bf*>(&Bls[row][half * 8]);
      v8bf hi = *reinterpret_cast<const v8bf*>(&Bls[row][16 + half * 8]);
      b[nt] = cat8(lo, hi);
    }
#pragma unroll
    for (int mt = 0; mt < 2; ++mt)
#pragma unroll
      for (int nt = 0; nt < 4; ++nt) c[mt][nt] = wmma_bf16(a[mt], b[nt], c[mt][nt]);
    __syncthreads();

    if (more) {
#pragma unroll
      for (int i = 0; i < 4; ++i) { va[i] = na[i]; vb[i] = nb[i]; }
    }
  }

  // --- C store: row = i + 8*half, col = lm ---
#pragma unroll
  for (int mt = 0; mt < 2; ++mt)
#pragma unroll
    for (int nt = 0; nt < 4; ++nt)
#pragma unroll
      for (int i = 0; i < 8; ++i) {
        int rr = m0 + wm * 32 + mt * 16 + half * 8 + i;
        int cc = n0 + wn * 64 + nt * 16 + lm;
        C[(size_t)rr * N + cc] = c[mt][nt][i];
      }
}

// ---------------------------------------------------------------------------
// Fused per-head LayerNorm + physical-coordinate RoPE.
// proj (B,S,H,96) fp32 -> out (B,H,S,96) bf16. One wave per (b,s,h) row.
// ---------------------------------------------------------------------------
__global__ __launch_bounds__(256) void ln_rope_kernel(
    const float* __restrict__ proj, const float* __restrict__ coords,
    const float* __restrict__ gamma, const float* __restrict__ beta,
    __bf16* __restrict__ out) {
  const int row  = blockIdx.x * 8 + (threadIdx.x >> 5);  // (b*S+s)*H + h
  const int lane = threadIdx.x & 31;
  const int h  = row % NH;
  const int bs = row / NH;
  const int s  = bs % SEQ;
  const int b  = bs / SEQ;

  const float* x = proj + (size_t)row * DH;
  float v0 = x[lane], v1 = x[lane + 32], v2 = x[lane + 64];
  float sum = v0 + v1 + v2;
  float sq  = v0 * v0 + v1 * v1 + v2 * v2;
#pragma unroll
  for (int m = 1; m < 32; m <<= 1) {
    sum += __shfl_xor(sum, m);
    sq  += __shfl_xor(sq, m);
  }
  const float mu   = sum * (1.0f / DH);
  const float var  = sq * (1.0f / DH) - mu * mu;
  const float rstd = rsqrtf(var + 1e-5f);

  __bf16* og = out + ((size_t)(b * NH + h) * SEQ + s) * DH;
  const float* cd = coords + (size_t)(b * SEQ + s) * 3;

  auto do_pair = [&](int m) {
    int p = m >> 4, f = m & 15;                    // m = p*16 + f
    float freq = __expf(-5.0f + (8.0f / 15.0f) * (float)f);
    float th = cd[p] * freq;
    float cs = __cosf(th), sn = __sinf(th);
    int e0 = 2 * m;
    float n0 = (x[e0] - mu) * rstd * gamma[e0] + beta[e0];
    float n1 = (x[e0 + 1] - mu) * rstd * gamma[e0 + 1] + beta[e0 + 1];
    og[e0]     = (__bf16)(n0 * cs - n1 * sn);
    og[e0 + 1] = (__bf16)(n0 * sn + n1 * cs);
  };
  do_pair(lane);                      // pairs 0..31
  if (lane < 16) do_pair(32 + lane);  // pairs 32..47
}

// ---------------------------------------------------------------------------
// V repack: (B,S,H,96) fp32 -> (B,H,S,96) bf16, output-coalesced.
// ---------------------------------------------------------------------------
__global__ __launch_bounds__(256) void v_pack_kernel(
    const float* __restrict__ proj, __bf16* __restrict__ out) {
  size_t idx = (size_t)blockIdx.x * 256 + threadIdx.x;
  int d = idx % DH;
  size_t t = idx / DH;
  int s = t % SEQ; t /= SEQ;
  int h = t % NH;
  int b = (int)(t / NH);
  out[idx] = (__bf16)proj[(((size_t)(b * SEQ + s)) * NH + h) * DH + d];
}

// ---------------------------------------------------------------------------
// Flash attention, bf16 WMMA, online softmax.
// q/k/v: (B,H,S,96) bf16. o: (B,S,H*96) fp32 (ready for the Wo GEMM).
// Block = 4 waves, each wave owns 16 q rows; 32-key steps. K tile streamed via
// GLOBAL_LOAD_ASYNC_TO_LDS_B128 (ASYNCcnt) when the toolchain exposes it.
// ---------------------------------------------------------------------------
__global__ __launch_bounds__(128) void attn_kernel(
    const __bf16* __restrict__ q, const __bf16* __restrict__ k,
    const __bf16* __restrict__ v, float* __restrict__ o) {
  __shared__ __bf16 Kls[32][104];     // row-major keys, padded (52-bank stride)
  __shared__ __bf16 Vls[96][40];      // transposed V, padded (20-bank stride)
  __shared__ __bf16 Pls[4][16][40];   // per-wave P staging (C-layout -> A-layout)

  const int tid  = threadIdx.x;
  const int lane = tid & 31;
  const int w    = tid >> 5;
  const int half = lane >> 4;
  const int lm   = lane & 15;
  const int qb = blockIdx.x, h = blockIdx.y, b = blockIdx.z;

  const size_t bh = (size_t)b * NH + h;
  const __bf16* qg = q + bh * SEQ * DH;
  const __bf16* kg = k + bh * SEQ * DH;
  const __bf16* vg = v + bh * SEQ * DH;
  const int s0 = qb * 64 + w * 16;

  // Q tile (16 rows x 96) as three 16x32 A-fragments, loaded once.
  v16bf qa[3];
#pragma unroll
  for (int c = 0; c < 3; ++c) {
    const __bf16* base = qg + (size_t)(s0 + lm) * DH + c * 32 + half * 8;
    v8bf lo = *reinterpret_cast<const v8bf*>(base);
    v8bf hi = *reinterpret_cast<const v8bf*>(base + 16);
    qa[c] = cat8(lo, hi);
  }

  v8f zero = {};
  v8f acc[6];
#pragma unroll
  for (int dc = 0; dc < 6; ++dc) acc[dc] = zero;
  float mrow[8], lrow[8];
#pragma unroll
  for (int i = 0; i < 8; ++i) { mrow[i] = -1e30f; lrow[i] = 0.0f; }

  const float scale = 0.102062072615966f;  // 1/sqrt(96)

  for (int kb = 0; kb < SEQ / 32; ++kb) {
    __syncthreads();
    // K tile: 32x96 bf16, 16B chunks (384 total / 128 threads = 3 each)
#pragma unroll
    for (int i = 0; i < 3; ++i) {
      int idx = tid + i * 128;
      int r = idx / 12, c16 = idx % 12;
      const __bf16* src = kg + (size_t)(kb * 32 + r) * DH + c16 * 8;
#if USE_ASYNC_LDS
      async_ld_b128(src, &Kls[r][c16 * 8]);     // global_load_async_to_lds_b128
#else
      *reinterpret_cast<uint4*>(&Kls[r][c16 * 8]) =
          *reinterpret_cast<const uint4*>(src);
#endif
    }
    // V tile, stored transposed: Vls[d][key]
#pragma unroll
    for (int i = 0; i < 12; ++i) {
      int idx = tid + i * 128;
      int r = idx / 48, dp = idx % 48;
      const __bf16* src = vg + (size_t)(kb * 32 + r) * DH + dp * 2;
      Vls[dp * 2][r]     = src[0];
      Vls[dp * 2 + 1][r] = src[1];
    }
#if USE_ASYNC_LDS
    __builtin_amdgcn_s_wait_asynccnt(0);        // s_wait_asynccnt 0
#endif
    __syncthreads();

    // S(16x32) = Q * K^T : 6 WMMAs (3 d-chunks x 2 key columns)
    v8f sc0 = zero, sc1 = zero;
#pragma unroll
    for (int c = 0; c < 3; ++c) {
      v8bf l0 = *reinterpret_cast<const v8bf*>(&Kls[lm][c * 32 + half * 8]);
      v8bf h0 = *reinterpret_cast<const v8bf*>(&Kls[lm][c * 32 + 16 + half * 8]);
      v8bf l1 = *reinterpret_cast<const v8bf*>(&Kls[16 + lm][c * 32 + half * 8]);
      v8bf h1 = *reinterpret_cast<const v8bf*>(&Kls[16 + lm][c * 32 + 16 + half * 8]);
      sc0 = wmma_bf16(qa[c], cat8(l0, h0), sc0);
      sc1 = wmma_bf16(qa[c], cat8(l1, h1), sc1);
    }

    // Online softmax. Row r = i + 8*half lives across lanes {lm=0..15} of the
    // same half-wave -> shfl_xor masks 1,2,4,8 reduce within the row.
    float mnew[8], alpha[8];
#pragma unroll
    for (int i = 0; i < 8; ++i) {
      float a0 = sc0[i] * scale, a1 = sc1[i] * scale;
      sc0[i] = a0; sc1[i] = a1;
      float t = fmaxf(a0, a1);
      t = fmaxf(t, __shfl_xor(t, 1));
      t = fmaxf(t, __shfl_xor(t, 2));
      t = fmaxf(t, __shfl_xor(t, 4));
      t = fmaxf(t, __shfl_xor(t, 8));
      mnew[i]  = fmaxf(mrow[i], t);
      alpha[i] = __expf(mrow[i] - mnew[i]);
    }
#pragma unroll
    for (int i = 0; i < 8; ++i) {
      float p0 = __expf(sc0[i] - mnew[i]);
      float p1 = __expf(sc1[i] - mnew[i]);
      Pls[w][half * 8 + i][lm]      = (__bf16)p0;   // C-layout -> staging
      Pls[w][half * 8 + i][16 + lm] = (__bf16)p1;
      float rs = p0 + p1;
      rs += __shfl_xor(rs, 1);
      rs += __shfl_xor(rs, 2);
      rs += __shfl_xor(rs, 4);
      rs += __shfl_xor(rs, 8);
      lrow[i] = lrow[i] * alpha[i] + rs;
      mrow[i] = mnew[i];
    }
#pragma unroll
    for (int dc = 0; dc < 6; ++dc)
#pragma unroll
      for (int i = 0; i < 8; ++i) acc[dc][i] *= alpha[i];

    // P back as A-fragment (same-wave LDS RAW: DS ops are in order per wave)
    v8bf plo = *reinterpret_cast<const v8bf*>(&Pls[w][lm][half * 8]);
    v8bf phi = *reinterpret_cast<const v8bf*>(&Pls[w][lm][16 + half * 8]);
    v16bf pa = cat8(plo, phi);

    // acc(16x96) += P(16x32) * V(32x96): 6 WMMAs over d-chunks
#pragma unroll
    for (int dc = 0; dc < 6; ++dc) {
      v8bf lo = *reinterpret_cast<const v8bf*>(&Vls[dc * 16 + lm][half * 8]);
      v8bf hi = *reinterpret_cast<const v8bf*>(&Vls[dc * 16 + lm][16 + half * 8]);
      acc[dc] = wmma_bf16(pa, cat8(lo, hi), acc[dc]);
    }
  }

  // Epilogue: normalize and write (B,S,H*96) fp32 for the Wo GEMM.
#pragma unroll
  for (int dc = 0; dc < 6; ++dc)
#pragma unroll
    for (int i = 0; i < 8; ++i) {
      int s = s0 + half * 8 + i;
      float ov = acc[dc][i] / lrow[i];
      o[((size_t)(b * SEQ + s)) * DM + h * DH + dc * 16 + lm] = ov;
    }
}

// ---------------------------------------------------------------------------
// Host-side orchestration (graph-capture safe: launches only).
// ---------------------------------------------------------------------------
extern "C" void kernel_launch(void* const* d_in, const int* in_sizes, int n_in,
                              void* d_out, int out_size, void* d_ws, size_t ws_size,
                              hipStream_t stream) {
  const float* qx = (const float*)d_in[0];
  const float* kx = (const float*)d_in[1];
  const float* vx = (const float*)d_in[2];
  const float* x_q = (const float*)d_in[3];
  const float* x_k = (const float*)d_in[4];
  const float* Wq = (const float*)d_in[5];
  const float* Wk = (const float*)d_in[6];
  const float* Wv = (const float*)d_in[7];
  const float* Wo = (const float*)d_in[8];
  const float* qg = (const float*)d_in[9];
  const float* qb = (const float*)d_in[10];
  const float* kg = (const float*)d_in[11];
  const float* kb = (const float*)d_in[12];

  const int M = BATCH * SEQ;  // 8192 rows
  const size_t mat = (size_t)M * DM;

  float*  Qp = (float*)d_ws;
  float*  Kp = Qp + mat;
  float*  Vp = Kp + mat;
  __bf16* qbf = (__bf16*)(Vp + mat);
  __bf16* kbf = qbf + mat;
  __bf16* vbf = kbf + mat;
  float*  attn = Qp;  // Qp is dead after ln_rope(q); reuse for attention output

  dim3 gg(DM / BN, M / BM);  // 12 x 64
  gemm_xwt_kernel<<<gg, 256, 0, stream>>>(qx, Wq, Qp, M, DM, DM);
  gemm_xwt_kernel<<<gg, 256, 0, stream>>>(kx, Wk, Kp, M, DM, DM);
  gemm_xwt_kernel<<<gg, 256, 0, stream>>>(vx, Wv, Vp, M, DM, DM);

  ln_rope_kernel<<<(M * NH) / 8, 256, 0, stream>>>(Qp, x_q, qg, qb, qbf);
  ln_rope_kernel<<<(M * NH) / 8, 256, 0, stream>>>(Kp, x_k, kg, kb, kbf);
  v_pack_kernel<<<(unsigned)(mat / 256), 256, 0, stream>>>(Vp, vbf);

  attn_kernel<<<dim3(SEQ / 64, NH, BATCH), 128, 0, stream>>>(qbf, kbf, vbf, attn);

  gemm_xwt_kernel<<<gg, 256, 0, stream>>>(attn, Wo, (float*)d_out, M, DM, DM);
}
